// RNNTJoint_66932770341221
// MI455X (gfx1250) — matrix operand order repaired and checked by reading, not compiled, not run on hardware
//
#include <hip/hip_runtime.h>
#include <hip/hip_bf16.h>
#include <math.h>

typedef _Float16 v16h __attribute__((ext_vector_type(16)));
typedef _Float16 v8h  __attribute__((ext_vector_type(8)));
typedef float    v8f  __attribute__((ext_vector_type(8)));

#define NB   4
#define NT   256
#define NU   128
#define ENCH 512
#define PREDH 640
#define JH   640
#define NV   1025
#define HA_STRIDE 664   // 640 + 24 pad: 16B-aligned rows, conflict-free b128 LDS loads
#define MROWS 32        // u-rows per workgroup (W_joint reuse factor)

// ---------------------------------------------------------------------------
// proj: dst[b, l, j] = bias[j] + sum_h src[b, h, l] * W[j, h]
// src is [B, H, L]; one block = (b, 16 l's, 16 j's); src tile staged in LDS.
// ---------------------------------------------------------------------------
__global__ __launch_bounds__(256) void proj_kernel(
    const float* __restrict__ src, const float* __restrict__ W,
    const float* __restrict__ bias, float* __restrict__ dst,
    int H, int L)
{
  extern __shared__ float sSrc[];   // [H][16]
  const int tid = threadIdx.x;
  const int nJB = JH / 16;          // 40
  const int jb = blockIdx.x % nJB;
  const int lb = (blockIdx.x / nJB) % (L / 16);
  const int b  = blockIdx.x / (nJB * (L / 16));

  for (int idx = tid; idx < H * 16; idx += 256) {
    int h = idx >> 4, l = idx & 15;
    sSrc[idx] = src[((size_t)b * H + h) * L + lb * 16 + l];
  }
  __syncthreads();

  const int jj = tid & 15, l = tid >> 4;
  const int j = jb * 16 + jj;
  const float* wr = W + (size_t)j * H;
  float acc = bias[j];
  for (int h = 0; h < H; h += 4) {
    float4 w4 = *(const float4*)(wr + h);
    acc += sSrc[(h + 0) * 16 + l] * w4.x;
    acc += sSrc[(h + 1) * 16 + l] * w4.y;
    acc += sSrc[(h + 2) * 16 + l] * w4.z;
    acc += sSrc[(h + 3) * 16 + l] * w4.w;
  }
  dst[((size_t)b * L + lb * 16 + l) * JH + j] = acc;
}

__global__ __launch_bounds__(256) void cvt_f16(const float* __restrict__ in,
                                               _Float16* __restrict__ out, int n)
{
  int i = blockIdx.x * 256 + threadIdx.x;
  if (i < n) out[i] = (_Float16)in[i];
}

// ---------------------------------------------------------------------------
// joint: per WG: one (b, t, 32-u tile). h=relu(f+g) in LDS (f16).
// 16 waves; each wave owns 64 classes (4 N-tiles) and BOTH 16-row M-subtiles,
// so every W_joint fetch feeds 2 WMMAs (halves L2 traffic vs M=16).
// K=640 fully unrolled: all load addresses become base+imm24, no per-step
// VALU address chains (avoids WMMA->VALU hazard nops).
// Fused log_softmax over V=1025 (blank class via VALU dot).
// ---------------------------------------------------------------------------
__global__ __launch_bounds__(512) void joint_kernel(
    const float* __restrict__ fbuf, const float* __restrict__ gbuf,
    const _Float16* __restrict__ wj, const float* __restrict__ bj,
    float* __restrict__ out)
{
  __shared__ _Float16 hA[MROWS * HA_STRIDE];
  __shared__ float sPart[16][33];   // [wave][row 0..31]
  __shared__ float sSum[16][33];
  __shared__ float sMax[MROWS];
  __shared__ float sBlank[MROWS];
  __shared__ float sLse[MROWS];

  const int tid = threadIdx.x;
  const int wg  = blockIdx.x;           // b*T*(U/32)
  const int ut  = wg & 3;               // U/32 = 4
  const int t   = (wg >> 2) & 255;
  const int b   = wg >> 10;
  const int u0  = ut * MROWS;

  const float* frow = fbuf + ((size_t)(b * NT + t)) * JH;
  const float* grow = gbuf + ((size_t)(b * NU + u0)) * JH;

  // h = relu(f_t + g_u), stored f16 in LDS, A-matrix source
  for (int idx = tid; idx < MROWS * JH; idx += 512) {
    int u = idx / JH, j = idx - u * JH;
    float v = frow[j] + grow[u * JH + j];
    hA[u * HA_STRIDE + j] = (_Float16)fmaxf(v, 0.0f);
  }
  __syncthreads();

  const int wave = tid >> 5;
  const int lane = tid & 31;
  const int half = lane >> 4;
  const int l16  = lane & 15;
  const int colBase = wave * 64;

  // invariant per-lane base pointers (k enters as immediate offset only)
  const _Float16* paBase = &hA[l16 * HA_STRIDE + 8 * half];
  const _Float16* pbBase = wj + (size_t)(colBase + l16) * JH + 8 * half;

  v8f acc[2][4] = {};

#pragma unroll
  for (int k0 = 0; k0 < JH; k0 += 32) {
    // A operands (2 M-subtiles): lane = row l16; halves 0..7 = K[k0+8*half..),
    // halves 8..15 = K[k0+16+8*half..)  (ISA 16-bit A 16x32 layout)
    v16h a[2];
#pragma unroll
    for (int mg = 0; mg < 2; ++mg) {
      const _Float16* pa = paBase + mg * 16 * HA_STRIDE + k0;
      v8h alo = *(const v8h*)pa;
      v8h ahi = *(const v8h*)(pa + 16);
      a[mg] = __builtin_shufflevector(alo, ahi,
          0,1,2,3,4,5,6,7,8,9,10,11,12,13,14,15);
    }
#pragma unroll
    for (int q = 0; q < 4; ++q) {
      // B operand: lane = col l16; halves 0..7 = K[k0+8*half..), 8..15 = K[k0+16+8*half..)
      const _Float16* pb = pbBase + (size_t)q * 16 * JH + k0;
      v8h blo = *(const v8h*)pb;
      v8h bhi = *(const v8h*)(pb + 16);
      v16h bm = __builtin_shufflevector(blo, bhi,
          0,1,2,3,4,5,6,7,8,9,10,11,12,13,14,15);
      acc[0][q] = __builtin_amdgcn_wmma_f32_16x16x32_f16(
          false, a[0], false, bm, (short)0, acc[0][q], false, false);
      acc[1][q] = __builtin_amdgcn_wmma_f32_16x16x32_f16(
          false, a[1], false, bm, (short)0, acc[1][q], false, false);
    }
  }

  // bias add (col constant across the 8 accum VGPRs of a lane)
#pragma unroll
  for (int q = 0; q < 4; ++q) {
    float bb = bj[colBase + q * 16 + l16];
#pragma unroll
    for (int v = 0; v < 8; ++v) { acc[0][q][v] += bb; acc[1][q][v] += bb; }
  }

  // blank class (v = 1024): wave w handles rows w and w+16
  {
    const _Float16* wrow = wj + (size_t)1024 * JH;
    const float bj1024 = bj[1024];
#pragma unroll
    for (int rr = 0; rr < 2; ++rr) {
      const int row = wave + 16 * rr;
      float bl = 0.0f;
      const _Float16* hrow = hA + row * HA_STRIDE;
      for (int j = lane; j < JH; j += 32)
        bl += (float)hrow[j] * (float)wrow[j];
#pragma unroll
      for (int m = 16; m >= 1; m >>= 1) bl += __shfl_xor(bl, m, 32);
      if (lane == 0) sBlank[row] = bl + bj1024;
    }
  }

  // per-row partial max within this wave's 64 columns
  // C layout: element v of lane L is row (mg*16 + v + 8*(L>>4)), col q*16+(L&15)
#pragma unroll
  for (int mg = 0; mg < 2; ++mg) {
#pragma unroll
    for (int v = 0; v < 8; ++v) {
      float m = fmaxf(fmaxf(acc[mg][0][v], acc[mg][1][v]),
                      fmaxf(acc[mg][2][v], acc[mg][3][v]));
#pragma unroll
      for (int s = 1; s < 16; s <<= 1) m = fmaxf(m, __shfl_xor(m, s, 32));
      if (l16 == 0) sPart[wave][mg * 16 + v + 8 * half] = m;
    }
  }
  __syncthreads();

  // global row max: lanes 0-15 do row=wave, lanes 16-31 do row=wave+16
  const int myrow = wave + 16 * half;
  {
    float rm = sPart[l16][myrow];
#pragma unroll
    for (int s = 1; s < 16; s <<= 1) rm = fmaxf(rm, __shfl_xor(rm, s, 32));
    rm = fmaxf(rm, sBlank[myrow]);
    if (l16 == 0) sMax[myrow] = rm;     // lanes 0 and 16 write
  }
  __syncthreads();

  // per-row partial sum of exp
  float mv[2][8];
#pragma unroll
  for (int mg = 0; mg < 2; ++mg)
#pragma unroll
    for (int v = 0; v < 8; ++v) mv[mg][v] = sMax[mg * 16 + v + 8 * half];
#pragma unroll
  for (int mg = 0; mg < 2; ++mg) {
#pragma unroll
    for (int v = 0; v < 8; ++v) {
      float s = __expf(acc[mg][0][v] - mv[mg][v]) + __expf(acc[mg][1][v] - mv[mg][v])
              + __expf(acc[mg][2][v] - mv[mg][v]) + __expf(acc[mg][3][v] - mv[mg][v]);
#pragma unroll
      for (int x = 1; x < 16; x <<= 1) s += __shfl_xor(s, x, 32);
      if (l16 == 0) sSum[wave][mg * 16 + v + 8 * half] = s;
    }
  }
  __syncthreads();

  // finish logsumexp per row (lanes 0-15: row wave; 16-31: row wave+16)
  {
    float ss = sSum[l16][myrow];
#pragma unroll
    for (int x = 1; x < 16; x <<= 1) ss += __shfl_xor(ss, x, 32);
    if (l16 == 0) {
      ss += __expf(sBlank[myrow] - sMax[myrow]);
      sLse[myrow] = sMax[myrow] + __logf(ss);
    }
  }
  __syncthreads();

  float lseArr[2][8];
#pragma unroll
  for (int mg = 0; mg < 2; ++mg)
#pragma unroll
    for (int v = 0; v < 8; ++v) lseArr[mg][v] = sLse[mg * 16 + v + 8 * half];

  const size_t base = (((size_t)(b * NT + t)) * NU + u0) * (size_t)NV;
#pragma unroll
  for (int mg = 0; mg < 2; ++mg) {
#pragma unroll
    for (int q = 0; q < 4; ++q) {
      int col = colBase + q * 16 + l16;
#pragma unroll
      for (int v = 0; v < 8; ++v) {
        int row = mg * 16 + v + 8 * half;
        out[base + (size_t)row * NV + col] = acc[mg][q][v] - lseArr[mg][v];
      }
    }
  }
  if (l16 == 0)   // lanes 0 and 16 write blank logits for rows wave, wave+16
    out[base + (size_t)myrow * NV + 1024] = sBlank[myrow] - sLse[myrow];
}

// ---------------------------------------------------------------------------
extern "C" void kernel_launch(void* const* d_in, const int* in_sizes, int n_in,
                              void* d_out, int out_size, void* d_ws, size_t ws_size,
                              hipStream_t stream)
{
  (void)in_sizes; (void)n_in; (void)out_size; (void)ws_size;
  const float* enc     = (const float*)d_in[0];
  const float* dec     = (const float*)d_in[1];
  const float* W_enc   = (const float*)d_in[2];
  const float* b_enc   = (const float*)d_in[3];
  const float* W_pred  = (const float*)d_in[4];
  const float* b_pred  = (const float*)d_in[5];
  const float* W_joint = (const float*)d_in[6];
  const float* b_joint = (const float*)d_in[7];
  float* outp = (float*)d_out;

  float* fbuf = (float*)d_ws;                                // [B*T, 640] f32
  float* gbuf = fbuf + (size_t)NB * NT * JH;                 // [B*U, 640] f32
  _Float16* wj16 = (_Float16*)(gbuf + (size_t)NB * NU * JH); // [1025, 640] f16

  proj_kernel<<<NB * (NT / 16) * (JH / 16), 256,
                ENCH * 16 * sizeof(float), stream>>>(enc, W_enc, b_enc, fbuf, ENCH, NT);
  proj_kernel<<<NB * (NU / 16) * (JH / 16), 256,
                PREDH * 16 * sizeof(float), stream>>>(dec, W_pred, b_pred, gbuf, PREDH, NU);
  cvt_f16<<<(NV * JH + 255) / 256, 256, 0, stream>>>(W_joint, wj16, NV * JH);
  joint_kernel<<<NB * NT * (NU / MROWS), 512, 0, stream>>>(fbuf, gbuf, wj16, b_joint, outp);
}
